// BoxNNDGCNN_31301721653636
// MI455X (gfx1250) — matrix operand order repaired and checked by reading, not compiled, not run on hardware
//
#include <hip/hip_runtime.h>
#include <stdint.h>

// ---------------- problem constants ----------------
#define NBATCH 64
#define NPG    512
#define NPTS   (NBATCH * NPG)     // 32768
#define KNBR   10
#define NMSG   (NPTS * KNBR)      // 327680
#define NEDGE  262144
#define MREP   4                  // 16-row strips per wave (64 rows/wave, 256 rows/block)

// ---------------- WMMA types ----------------
typedef __attribute__((ext_vector_type(16))) __bf16 v16bf;
typedef __attribute__((ext_vector_type(8)))  float  v8f;

union BfFrag { v16bf v; uint4 q[2]; unsigned short u[16]; };

__device__ __forceinline__ unsigned short f2bf(float f) {
  unsigned int u = __float_as_uint(f);
  u += 0x7FFFu + ((u >> 16) & 1u);           // round-to-nearest-even
  return (unsigned short)(u >> 16);
}

// ---------------- fragment-order swizzles (ISA 7.12.2 layouts) ----------------
// A (16-bit, 16x32): lane<16: e<8->K=e, e>=8->K=e+8 ; lane>=16: e<8->K=e+8, e>=8->K=e+16
// Forward map for element (row, kk) of an M x Kp panel:
__device__ __forceinline__ size_t a_swz_idx(size_t row, int kk, int Kp) {
  size_t m16 = row >> 4;
  int rl   = (int)(row & 15);
  int kblk = kk >> 5;
  int kl   = kk & 31;
  int half = (kl >> 3) & 1;                       // 0:[0,8)|[16,24)  1:[8,16)|[24,32)
  int e    = (kl & 7) + ((kl >= 16) ? 8 : 0);
  int lane = half * 16 + rl;
  return ((m16 * (size_t)(Kp >> 5) + kblk) * 32 + lane) * 16 + e;
}
// B (16-bit, 32x16): lane<16: e<8->K=e, e>=8->K=e+8 ; lane>=16: e<8->K=e+8, e>=8->K=e+16 ; col = lane&15
__device__ __forceinline__ size_t b_swz_idx(int k, size_t n, int Kp) {
  size_t nblk = n >> 4;
  int col  = (int)(n & 15);
  int kblk = k >> 5;
  int kl   = k & 31;
  int lane = ((kl >> 3) & 1) * 16 + col;
  int e    = (kl & 7) + ((kl >= 16) ? 8 : 0);
  return ((nblk * (size_t)(Kp >> 5) + kblk) * 32 + lane) * 16 + e;
}

// =====================================================================
// GEMM: C[M x ldc] = act(A[M x Kp] @ W[Kp x Np] + bias), operands bf16 in
// fragment order, f32 accumulate. Block = 128 thr (4 waves); each wave owns
// 64 rows x 64 cols (MREP strips x 4 col-blocks -> 16 wmma per K-step).
// No LDS, no syncs: A/B fragments are direct 2x b128 global loads.
// Requires M % 256 == 0 (true for all call sites). act: 0 none, 1 leaky, 2 relu
// =====================================================================
__global__ __launch_bounds__(128) void gemm_bf16_wmma(
    const unsigned short* __restrict__ A,
    const unsigned short* __restrict__ W,
    const float* __restrict__ bias,
    float* __restrict__ C,
    int M, int Kp, int Np, int Nvalid, int ldc, int act)
{
  const int tid  = threadIdx.x;
  const int wave = tid >> 5;
  const int lane = tid & 31;
  const int KB   = Kp >> 5;
  const size_t m16base = (size_t)blockIdx.x * (4 * MREP) + wave * MREP;
  const int n0 = blockIdx.y * 64;
  const size_t nblk0 = (size_t)(n0 >> 4);

  v8f acc[MREP][4];
  #pragma unroll
  for (int mi = 0; mi < MREP; ++mi)
    #pragma unroll
    for (int nb = 0; nb < 4; ++nb)
      #pragma unroll
      for (int r = 0; r < 8; ++r) acc[mi][nb][r] = 0.0f;

  const uint4* __restrict__ Ag = (const uint4*)A;   // 16B units; lane frag = 2
  const uint4* __restrict__ Wg = (const uint4*)W;

  for (int kblk = 0; kblk < KB; ++kblk) {
    BfFrag fa[MREP];
    #pragma unroll
    for (int mi = 0; mi < MREP; ++mi) {
      const size_t base = (((m16base + mi) * KB + kblk) * 32 + lane) * 2;
      fa[mi].q[0] = Ag[base];
      fa[mi].q[1] = Ag[base + 1];
    }
    // prefetch next A K-block (global_prefetch_b8); workspace extends past panel
    __builtin_prefetch(&Ag[(((m16base) * KB + kblk + 1) * 32 + lane) * 2], 0, 0);

    #pragma unroll
    for (int nb = 0; nb < 4; ++nb) {
      const size_t bbase = (((nblk0 + nb) * KB + kblk) * 32 + lane) * 2;
      BfFrag fb;
      fb.q[0] = Wg[bbase];
      fb.q[1] = Wg[bbase + 1];
      #pragma unroll
      for (int mi = 0; mi < MREP; ++mi)
        acc[mi][nb] = __builtin_amdgcn_wmma_f32_16x16x32_bf16(
            false, fa[mi].v, false, fb.v, (short)0, acc[mi][nb], false, false);
    }
  }

  // C/D layout: lane<16 -> M=r, lane>=16 -> M=8+r ; N = lane&15
  #pragma unroll
  for (int mi = 0; mi < MREP; ++mi) {
    const size_t rbase = (m16base + mi) * 16 + ((lane < 16) ? 0 : 8);
    #pragma unroll
    for (int nb = 0; nb < 4; ++nb) {
      const int col = n0 + nb * 16 + (lane & 15);
      const float bv = (bias != nullptr && col < Nvalid) ? bias[col] : 0.0f;
      #pragma unroll
      for (int r = 0; r < 8; ++r) {
        float y = acc[mi][nb][r] + bv;
        if (act == 1)      y = (y > 0.0f) ? y : 0.01f * y;
        else if (act == 2) y = (y > 0.0f) ? y : 0.0f;
        if (col < Nvalid)
          C[(rbase + r) * ldc + col] = y;
      }
    }
  }
}

// =====================================================================
// Per-batch Gram: S[b] = X[b](512 x Cp) @ X[b]^T. Xa = A-swizzled copy,
// Xbt = B-swizzled copy (both global-node indexed). grid (2, 8, 64).
// =====================================================================
__global__ __launch_bounds__(128) void gram_bf16_wmma(
    const unsigned short* __restrict__ Xa,
    const unsigned short* __restrict__ Xbt,
    int Cp, float* __restrict__ S)
{
  const int tid  = threadIdx.x;
  const int wave = tid >> 5;
  const int lane = tid & 31;
  const int b  = blockIdx.z;
  const int KB = Cp >> 5;
  const size_t m16base = (size_t)b * 32 + blockIdx.x * (4 * MREP) + wave * MREP;
  const int n0 = blockIdx.y * 64;                     // col within batch
  const size_t nblk0 = (size_t)b * 32 + (n0 >> 4);

  v8f acc[MREP][4];
  #pragma unroll
  for (int mi = 0; mi < MREP; ++mi)
    #pragma unroll
    for (int nb = 0; nb < 4; ++nb)
      #pragma unroll
      for (int r = 0; r < 8; ++r) acc[mi][nb][r] = 0.0f;

  const uint4* __restrict__ Ag = (const uint4*)Xa;
  const uint4* __restrict__ Bg = (const uint4*)Xbt;

  for (int kblk = 0; kblk < KB; ++kblk) {
    BfFrag fa[MREP];
    #pragma unroll
    for (int mi = 0; mi < MREP; ++mi) {
      const size_t base = (((m16base + mi) * KB + kblk) * 32 + lane) * 2;
      fa[mi].q[0] = Ag[base];
      fa[mi].q[1] = Ag[base + 1];
    }
    #pragma unroll
    for (int nb = 0; nb < 4; ++nb) {
      const size_t bbase = (((nblk0 + nb) * KB + kblk) * 32 + lane) * 2;
      BfFrag fb;
      fb.q[0] = Bg[bbase];
      fb.q[1] = Bg[bbase + 1];
      #pragma unroll
      for (int mi = 0; mi < MREP; ++mi)
        acc[mi][nb] = __builtin_amdgcn_wmma_f32_16x16x32_bf16(
            false, fa[mi].v, false, fb.v, (short)0, acc[mi][nb], false, false);
    }
  }

  float* Sb = S + (size_t)b * NPG * NPG;
  const size_t mloc = (size_t)blockIdx.x * (4 * MREP) + wave * MREP;  // strip within batch
  #pragma unroll
  for (int mi = 0; mi < MREP; ++mi) {
    const size_t rbase = (mloc + mi) * 16 + ((lane < 16) ? 0 : 8);
    #pragma unroll
    for (int nb = 0; nb < 4; ++nb) {
      const int col = n0 + nb * 16 + (lane & 15);
      #pragma unroll
      for (int r = 0; r < 8; ++r)
        Sb[(rbase + r) * NPG + col] = acc[mi][nb][r];
    }
  }
}

// ===================== support kernels =====================

// deterministic column stats, stage 1: per-512-row-chunk partial sums
__global__ void colstats_kernel(const float* __restrict__ X, int M, int C,
                                float* __restrict__ part)
{
  const int c = threadIdx.x;            // blockDim.x == C
  const int b = blockIdx.x;
  int r0 = b * 512;
  int r1 = r0 + 512; if (r1 > M) r1 = M;
  float s = 0.0f, s2 = 0.0f;
  for (int r = r0; r < r1; ++r) {
    float v = X[(size_t)r * C + c];
    s += v; s2 += v * v;
  }
  part[(size_t)b * 2 * C + c]     = s;
  part[(size_t)b * 2 * C + C + c] = s2;
}

// stage 2: reduce partials -> mean | rstd
__global__ void finstats_kernel(const float* __restrict__ part, int nb, int M, int C,
                                float* __restrict__ mrs)
{
  const int c = threadIdx.x;
  if (c >= C) return;
  float s = 0.0f, s2 = 0.0f;
  for (int b = 0; b < nb; ++b) {
    s  += part[(size_t)b * 2 * C + c];
    s2 += part[(size_t)b * 2 * C + C + c];
  }
  float m = s / (float)M;
  float v = s2 / (float)M - m * m;
  mrs[c]     = m;
  mrs[C + c] = rsqrtf(v + 1e-5f);
}

// in-place BN apply: h = (h-m)*rstd [* gamma + beta]
__global__ void bn_apply_kernel(float* __restrict__ H, size_t total, int C,
                                const float* __restrict__ mrs,
                                const float* __restrict__ gamma,
                                const float* __restrict__ beta)
{
  size_t g = (size_t)blockIdx.x * blockDim.x + threadIdx.x;
  if (g >= total) return;
  int c = (int)(g % (size_t)C);
  float y = (H[g] - mrs[c]) * mrs[C + c];
  if (gamma) y = y * gamma[c] + beta[c];
  H[g] = y;
}

// f32 -> bf16 with optional input-BN, zero K-pad, scatter to A-fragment order
__global__ void cvt_a_swz_kernel(const float* __restrict__ X, int C, int Kp, size_t M,
                                 const float* __restrict__ mrs,
                                 unsigned short* __restrict__ out)
{
  size_t g = (size_t)blockIdx.x * blockDim.x + threadIdx.x;
  size_t total = M * (size_t)Kp;
  if (g >= total) return;
  size_t row = g / (size_t)Kp;
  int kk = (int)(g % (size_t)Kp);
  unsigned short o = 0;
  if (kk < C) {
    float v = X[row * C + kk];
    if (mrs) v = (v - mrs[kk]) * mrs[C + kk];
    o = f2bf(v);
  }
  out[a_swz_idx(row, kk, Kp)] = o;
}

// f32 -> bf16 scatter to B-fragment order (node rows as B columns, for Gram)
__global__ void cvt_b_swz_kernel(const float* __restrict__ X, int C, int Kp, size_t M,
                                 unsigned short* __restrict__ out)
{
  size_t g = (size_t)blockIdx.x * blockDim.x + threadIdx.x;
  size_t total = M * (size_t)Kp;
  if (g >= total) return;
  size_t n = g / (size_t)Kp;
  int k = (int)(g % (size_t)Kp);
  unsigned short o = (k < C) ? f2bf(X[n * C + k]) : (unsigned short)0;
  out[b_swz_idx(k, n, Kp)] = o;
}

// concat two f32 panels -> bf16 A-fragment order (pad to Kp)
__global__ void concat2_swz_kernel(const float* __restrict__ A, int Ca,
                                   const float* __restrict__ Bp, int Cb,
                                   int Kp, size_t M, unsigned short* __restrict__ out)
{
  size_t g = (size_t)blockIdx.x * blockDim.x + threadIdx.x;
  size_t total = M * (size_t)Kp;
  if (g >= total) return;
  size_t row = g / (size_t)Kp;
  int c = (int)(g % (size_t)Kp);
  unsigned short o = 0;
  if (c < Ca)           o = f2bf(A[row * Ca + c]);
  else if (c < Ca + Cb) o = f2bf(Bp[row * Cb + (c - Ca)]);
  out[a_swz_idx(row, c, Kp)] = o;
}

// row squared-norms
__global__ void rowsq_kernel(const float* __restrict__ X, int C, int M,
                             float* __restrict__ sq)
{
  int n = blockIdx.x * blockDim.x + threadIdx.x;
  if (n >= M) return;
  float s = 0.0f;
  for (int c = 0; c < C; ++c) { float v = X[(size_t)n * C + c]; s += v * v; }
  sq[n] = s;
}

// per-row top-KNBR smallest d2 (self included, matching top_k(-d2))
__global__ void knn_kernel(const float* __restrict__ S, const float* __restrict__ sq,
                           int* __restrict__ idxout)
{
  int g = blockIdx.x * blockDim.x + threadIdx.x;
  if (g >= NPTS) return;
  int b = g / NPG;
  int i = g % NPG;
  const float* Srow = S + ((size_t)b * NPG + i) * NPG;
  const float* sqb  = sq + (size_t)b * NPG;
  float sqi = sqb[i];
  float bd[KNBR]; int bi[KNBR];
  #pragma unroll
  for (int k = 0; k < KNBR; ++k) { bd[k] = 3.0e38f; bi[k] = 0; }
  for (int j = 0; j < NPG; ++j) {
    float d = sqi + sqb[j] - 2.0f * Srow[j];
    if (d < bd[KNBR - 1]) {
      int pos = KNBR - 1;
      while (pos > 0 && bd[pos - 1] > d) {
        bd[pos] = bd[pos - 1]; bi[pos] = bi[pos - 1]; --pos;
      }
      bd[pos] = d; bi[pos] = j;
    }
  }
  #pragma unroll
  for (int k = 0; k < KNBR; ++k)
    idxout[(size_t)g * KNBR + k] = b * NPG + bi[k];
}

// build edge-conv message inputs [x_i, x_j - x_i] -> bf16 A-fragment order
__global__ void edge_feat_swz_kernel(const float* __restrict__ X,
                                     const int* __restrict__ idx,
                                     int C, int Kp, unsigned short* __restrict__ out)
{
  size_t g = (size_t)blockIdx.x * blockDim.x + threadIdx.x;
  size_t total = (size_t)NMSG * Kp;
  if (g >= total) return;
  size_t row = g / (size_t)Kp;
  int c = (int)(g % (size_t)Kp);
  size_t n = row / KNBR;
  unsigned short o = 0;
  if (c < C) {
    o = f2bf(X[n * C + c]);
  } else if (c < 2 * C) {
    int j  = idx[row];
    int cc = c - C;
    o = f2bf(X[(size_t)j * C + cc] - X[n * C + cc]);
  }
  out[a_swz_idx(row, c, Kp)] = o;
}

// max over K neighbors: (NPTS*KNBR, C) -> (NPTS, C)
__global__ void maxk_kernel(const float* __restrict__ msg, int C,
                            float* __restrict__ out)
{
  size_t g = (size_t)blockIdx.x * blockDim.x + threadIdx.x;
  size_t total = (size_t)NPTS * C;
  if (g >= total) return;
  size_t n = g / (size_t)C;
  int c = (int)(g % (size_t)C);
  float m = -3.0e38f;
  #pragma unroll
  for (int k = 0; k < KNBR; ++k)
    m = fmaxf(m, msg[((size_t)n * KNBR + k) * C + c]);
  out[g] = m;
}

// elementwise max
__global__ void max2_kernel(const float* __restrict__ A, const float* __restrict__ B,
                            float* __restrict__ out, size_t n)
{
  size_t g = (size_t)blockIdx.x * blockDim.x + threadIdx.x;
  if (g < n) out[g] = fmaxf(A[g], B[g]);
}

// edge classifier input: [nf[row], nf[col], edge_attr, pad16] -> bf16 A-frag order
__global__ void edge_in_swz_kernel(const float* __restrict__ nf,
                                   const int* __restrict__ ei,
                                   const float* __restrict__ ea,
                                   unsigned short* __restrict__ out)
{
  size_t g = (size_t)blockIdx.x * blockDim.x + threadIdx.x;
  size_t total = (size_t)NEDGE * 288;
  if (g >= total) return;
  size_t e = g / 288;
  int c = (int)(g % 288);
  unsigned short o = 0;
  if (c < 128) {
    int r = ei[e];
    o = f2bf(nf[(size_t)r * 128 + c]);
  } else if (c < 256) {
    int cl = ei[(size_t)NEDGE + e];
    o = f2bf(nf[(size_t)cl * 128 + (c - 128)]);
  } else if (c < 272) {
    o = f2bf(ea[e * 16 + (c - 256)]);
  }
  out[a_swz_idx(e, c, 288)] = o;
}

// weight f32 (K x N) -> bf16 B-fragment order (Kp x Np), zero-padded
__global__ void cvtw_swz_kernel(const float* __restrict__ W, int K, int Nsrc,
                                int Kp, int Np, unsigned short* __restrict__ out)
{
  size_t g = (size_t)blockIdx.x * blockDim.x + threadIdx.x;
  size_t total = (size_t)Kp * Np;
  if (g >= total) return;
  int k = (int)(g / (size_t)Np);
  int n = (int)(g % (size_t)Np);
  unsigned short o = (k < K && n < Nsrc) ? f2bf(W[(size_t)k * Nsrc + n]) : (unsigned short)0;
  out[b_swz_idx(k, (size_t)n, Kp)] = o;
}

// ===================== host orchestration =====================
extern "C" void kernel_launch(void* const* d_in, const int* in_sizes, int n_in,
                              void* d_out, int out_size, void* d_ws, size_t ws_size,
                              hipStream_t stream)
{
  (void)in_sizes; (void)n_in; (void)out_size; (void)ws_size;

  const float* x     = (const float*)d_in[0];
  const float* nnf   = (const float*)d_in[1];
  const float* ea    = (const float*)d_in[2];
  const int*   ei    = (const int*)  d_in[3];
  const float* ne_w1 = (const float*)d_in[4];  const float* ne_b1 = (const float*)d_in[5];
  const float* ne_w2 = (const float*)d_in[6];  const float* ne_b2 = (const float*)d_in[7];
  const float* fe_w1 = (const float*)d_in[8];  const float* fe_b1 = (const float*)d_in[9];
  const float* fe_w2 = (const float*)d_in[10]; const float* fe_b2 = (const float*)d_in[11];
  const float* fu_w  = (const float*)d_in[12]; const float* fu_b  = (const float*)d_in[13];
  const float* c1_w1 = (const float*)d_in[14]; const float* c1_b1 = (const float*)d_in[15];
  const float* c1_g1 = (const float*)d_in[16]; const float* c1_be1= (const float*)d_in[17];
  const float* c1_w2 = (const float*)d_in[18]; const float* c1_b2 = (const float*)d_in[19];
  const float* c1_g2 = (const float*)d_in[20]; const float* c1_be2= (const float*)d_in[21];
  const float* c1_w3 = (const float*)d_in[22]; const float* c1_b3 = (const float*)d_in[23];
  const float* c1_g3 = (const float*)d_in[24]; const float* c1_be3= (const float*)d_in[25];
  const float* c2_w  = (const float*)d_in[26]; const float* c2_b  = (const float*)d_in[27];
  const float* c2_g  = (const float*)d_in[28]; const float* c2_be = (const float*)d_in[29];
  const float* go_w  = (const float*)d_in[30]; const float* go_b  = (const float*)d_in[31];
  const float* go_g  = (const float*)d_in[32]; const float* go_be = (const float*)d_in[33];
  const float* nc_w1 = (const float*)d_in[34]; const float* nc_b1 = (const float*)d_in[35];
  const float* nc_w2 = (const float*)d_in[36]; const float* nc_b2 = (const float*)d_in[37];
  const float* ec_w1 = (const float*)d_in[38]; const float* ec_b1 = (const float*)d_in[39];
  const float* ec_w2 = (const float*)d_in[40]; const float* ec_b2 = (const float*)d_in[41];

  float* out = (float*)d_out;

  // ---- workspace carve-up ----
  char* ws = (char*)d_ws;
  size_t off = 0;
  auto alloc = [&](size_t bytes) -> void* {
    void* p = ws + off;
    off += (bytes + 255) & ~(size_t)255;
    return p;
  };

  float* part = (float*)alloc((size_t)640 * 256 * 4);   // stats partials
  float* mrs  = (float*)alloc(256 * 4);                 // mean | rstd

  unsigned short* wne1 = (unsigned short*)alloc((size_t)32  * 128 * 2);
  unsigned short* wne2 = (unsigned short*)alloc((size_t)128 * 128 * 2);
  unsigned short* wfe1 = (unsigned short*)alloc((size_t)64  * 128 * 2);
  unsigned short* wfe2 = (unsigned short*)alloc((size_t)128 * 128 * 2);
  unsigned short* wfu  = (unsigned short*)alloc((size_t)256 * 128 * 2);
  unsigned short* wc11 = (unsigned short*)alloc((size_t)256 * 64  * 2);
  unsigned short* wc12 = (unsigned short*)alloc((size_t)64  * 64  * 2);
  unsigned short* wc13 = (unsigned short*)alloc((size_t)64  * 64  * 2);
  unsigned short* wc2  = (unsigned short*)alloc((size_t)128 * 128 * 2);
  unsigned short* wgo  = (unsigned short*)alloc((size_t)192 * 128 * 2);
  unsigned short* wnc1 = (unsigned short*)alloc((size_t)128 * 128 * 2);
  unsigned short* wnc2 = (unsigned short*)alloc((size_t)128 * 64  * 2);  // Np padded to 64
  unsigned short* wec1 = (unsigned short*)alloc((size_t)288 * 128 * 2);
  unsigned short* wec2 = (unsigned short*)alloc((size_t)128 * 64  * 2);  // Np padded to 64

  unsigned short* BA = (unsigned short*)alloc((size_t)NMSG * 256 * 2);  // A staging (max user)
  unsigned short* XB = (unsigned short*)alloc((size_t)NPTS * 128 * 2);  // Gram B-swizzled X
  float* M1  = (float*)alloc((size_t)NMSG * 128 * 4);                   // message / edge hidden
  float* G   = (float*)alloc((size_t)NBATCH * NPG * NPG * 4);           // Gram output
  float* Fbox= (float*)alloc((size_t)NPTS * 128 * 4);
  float* Fnn = (float*)alloc((size_t)NPTS * 128 * 4);
  float* Ffus= (float*)alloc((size_t)NPTS * 128 * 4);
  float* Ftmp= (float*)alloc((size_t)NPTS * 128 * 4);
  float* Fnf = (float*)alloc((size_t)NPTS * 128 * 4);
  float* Fg1 = (float*)alloc((size_t)NPTS * 64  * 4);
  float* Fg2 = (float*)alloc((size_t)NPTS * 128 * 4);
  float* SQ  = (float*)alloc((size_t)NPTS * 4);
  int*   IDX = (int*)  alloc((size_t)NMSG * 4);

  auto cdiv = [](size_t a, size_t b) { return (a + b - 1) / b; };

  auto cvtw = [&](const float* Wp, int K, int Nsrc, int Kp, int Np, unsigned short* dst) {
    size_t total = (size_t)Kp * Np;
    cvtw_swz_kernel<<<dim3((unsigned)cdiv(total, 256)), 256, 0, stream>>>(Wp, K, Nsrc, Kp, Np, dst);
  };
  cvtw(ne_w1, 32, 128, 32, 128, wne1);
  cvtw(ne_w2, 128,128, 128,128, wne2);
  cvtw(fe_w1, 64, 128, 64, 128, wfe1);
  cvtw(fe_w2, 128,128, 128,128, wfe2);
  cvtw(fu_w,  256,128, 256,128, wfu);
  cvtw(c1_w1, 256,64,  256,64,  wc11);
  cvtw(c1_w2, 64, 64,  64, 64,  wc12);
  cvtw(c1_w3, 64, 64,  64, 64,  wc13);
  cvtw(c2_w,  128,128, 128,128, wc2);
  cvtw(go_w,  192,128, 192,128, wgo);
  cvtw(nc_w1, 128,128, 128,128, wnc1);
  cvtw(nc_w2, 128,10,  128,64,  wnc2);
  cvtw(ec_w1, 272,128, 288,128, wec1);
  cvtw(ec_w2, 128,4,   128,64,  wec2);

  auto stats = [&](const float* X, int M, int C) {
    int nb = (int)cdiv((size_t)M, 512);
    colstats_kernel<<<dim3((unsigned)nb), C, 0, stream>>>(X, M, C, part);
    finstats_kernel<<<1, C, 0, stream>>>(part, nb, M, C, mrs);
  };
  auto bnapply = [&](float* H, size_t M, int C, const float* g, const float* be) {
    size_t total = M * (size_t)C;
    bn_apply_kernel<<<dim3((unsigned)cdiv(total, 256)), 256, 0, stream>>>(H, total, C, mrs, g, be);
  };
  auto cvtA = [&](const float* X, int C, int Kp, size_t M, const float* mr, unsigned short* dst) {
    size_t total = M * (size_t)Kp;
    cvt_a_swz_kernel<<<dim3((unsigned)cdiv(total, 256)), 256, 0, stream>>>(X, C, Kp, M, mr, dst);
  };
  auto cvtB = [&](const float* X, int C, int Kp, size_t M, unsigned short* dst) {
    size_t total = M * (size_t)Kp;
    cvt_b_swz_kernel<<<dim3((unsigned)cdiv(total, 256)), 256, 0, stream>>>(X, C, Kp, M, dst);
  };
  auto gemm = [&](const unsigned short* A, const unsigned short* Wp, const float* bias,
                  float* Cc, int M, int Kp, int Np, int Nvalid, int ldc, int act) {
    dim3 grid((unsigned)(M / 256), (unsigned)cdiv((size_t)Nvalid, 64));
    gemm_bf16_wmma<<<grid, 128, 0, stream>>>(A, Wp, bias, Cc, M, Kp, Np, Nvalid, ldc, act);
  };

  // ================= node encoders =================
  stats(x, NPTS, 32);
  cvtA(x, 32, 32, NPTS, mrs, BA);
  gemm(BA, wne1, ne_b1, Ftmp, NPTS, 32, 128, 128, 128, 1);
  cvtA(Ftmp, 128, 128, NPTS, nullptr, BA);
  gemm(BA, wne2, ne_b2, Fbox, NPTS, 128, 128, 128, 128, 1);

  stats(nnf, NPTS, 64);
  cvtA(nnf, 64, 64, NPTS, mrs, BA);
  gemm(BA, wfe1, fe_b1, Ftmp, NPTS, 64, 128, 128, 128, 1);
  cvtA(Ftmp, 128, 128, NPTS, nullptr, BA);
  gemm(BA, wfe2, fe_b2, Fnn, NPTS, 128, 128, 128, 128, 1);

  {
    size_t total = (size_t)NPTS * 256;
    concat2_swz_kernel<<<dim3((unsigned)cdiv(total, 256)), 256, 0, stream>>>(
        Fbox, 128, Fnn, 128, 256, (size_t)NPTS, BA);
  }
  gemm(BA, wfu, fu_b, Ffus, NPTS, 256, 128, 128, 128, 1);

  // ================= edge conv 1 (fusion, C=128) =================
  cvtA(Ffus, 128, 128, NPTS, nullptr, BA);
  cvtB(Ffus, 128, 128, NPTS, XB);
  gram_bf16_wmma<<<dim3(2, 8, NBATCH), 128, 0, stream>>>(BA, XB, 128, G);
  rowsq_kernel<<<dim3((unsigned)cdiv(NPTS, 256)), 256, 0, stream>>>(Ffus, 128, NPTS, SQ);
  knn_kernel<<<dim3((unsigned)cdiv(NPTS, 128)), 128, 0, stream>>>(G, SQ, IDX);
  {
    size_t total = (size_t)NMSG * 256;
    edge_feat_swz_kernel<<<dim3((unsigned)cdiv(total, 256)), 256, 0, stream>>>(Ffus, IDX, 128, 256, BA);
  }
  gemm(BA, wc11, c1_b1, M1, NMSG, 256, 64, 64, 64, 1);
  stats(M1, NMSG, 64);  bnapply(M1, NMSG, 64, c1_g1, c1_be1);
  cvtA(M1, 64, 64, NMSG, nullptr, BA);
  gemm(BA, wc12, c1_b2, M1, NMSG, 64, 64, 64, 64, 1);
  stats(M1, NMSG, 64);  bnapply(M1, NMSG, 64, c1_g2, c1_be2);
  cvtA(M1, 64, 64, NMSG, nullptr, BA);
  gemm(BA, wc13, c1_b3, M1, NMSG, 64, 64, 64, 64, 1);
  stats(M1, NMSG, 64);  bnapply(M1, NMSG, 64, c1_g3, c1_be3);
  {
    size_t total = (size_t)NPTS * 64;
    maxk_kernel<<<dim3((unsigned)cdiv(total, 256)), 256, 0, stream>>>(M1, 64, Fg1);
  }

  // ================= edge conv 2 (g1, C=64) =================
  cvtA(Fg1, 64, 64, NPTS, nullptr, BA);
  cvtB(Fg1, 64, 64, NPTS, XB);
  gram_bf16_wmma<<<dim3(2, 8, NBATCH), 128, 0, stream>>>(BA, XB, 64, G);
  rowsq_kernel<<<dim3((unsigned)cdiv(NPTS, 256)), 256, 0, stream>>>(Fg1, 64, NPTS, SQ);
  knn_kernel<<<dim3((unsigned)cdiv(NPTS, 128)), 128, 0, stream>>>(G, SQ, IDX);
  {
    size_t total = (size_t)NMSG * 128;
    edge_feat_swz_kernel<<<dim3((unsigned)cdiv(total, 256)), 256, 0, stream>>>(Fg1, IDX, 64, 128, BA);
  }
  gemm(BA, wc2, c2_b, M1, NMSG, 128, 128, 128, 128, 1);
  stats(M1, NMSG, 128);  bnapply(M1, NMSG, 128, c2_g, c2_be);
  {
    size_t total = (size_t)NPTS * 128;
    maxk_kernel<<<dim3((unsigned)cdiv(total, 256)), 256, 0, stream>>>(M1, 128, Fg2);
  }

  // ================= graph out + node features =================
  {
    size_t total = (size_t)NPTS * 192;
    concat2_swz_kernel<<<dim3((unsigned)cdiv(total, 256)), 256, 0, stream>>>(
        Fg1, 64, Fg2, 128, 192, (size_t)NPTS, BA);
  }
  gemm(BA, wgo, go_b, Ftmp, NPTS, 192, 128, 128, 128, 1);
  stats(Ftmp, NPTS, 128);  bnapply(Ftmp, NPTS, 128, go_g, go_be);
  {
    size_t total = (size_t)NPTS * 128;
    max2_kernel<<<dim3((unsigned)cdiv(total, 256)), 256, 0, stream>>>(Fbox, Ftmp, Fnf, total);
  }

  // ================= node classifier -> d_out[0 : 327680] =================
  cvtA(Fnf, 128, 128, NPTS, nullptr, BA);
  gemm(BA, wnc1, nc_b1, Ftmp, NPTS, 128, 128, 128, 128, 1);
  cvtA(Ftmp, 128, 128, NPTS, nullptr, BA);
  gemm(BA, wnc2, nc_b2, out, NPTS, 128, 64, 10, 10, 2);

  // ================= edge classifier -> d_out[327680 : +1048576] =================
  {
    size_t total = (size_t)NEDGE * 288;
    edge_in_swz_kernel<<<dim3((unsigned)cdiv(total, 256)), 256, 0, stream>>>(Fnf, ei, ea, BA);
  }
  gemm(BA, wec1, ec_b1, M1, NEDGE, 288, 128, 128, 128, 1);
  cvtA(M1, 128, 128, NEDGE, nullptr, BA);
  gemm(BA, wec2, ec_b2, out + (size_t)NPTS * 10, NEDGE, 128, 64, 4, 4, 2);
}
